// GreenGNN_11441792877243
// MI455X (gfx1250) — matrix operand order repaired and checked by reading, not compiled, not run on hardware
//
#include <hip/hip_runtime.h>

typedef __bf16 bf16_t;
typedef __attribute__((ext_vector_type(16))) __bf16 v16bf;
typedef __attribute__((ext_vector_type(8)))  float  v8f;

#define NN_NODES 50000
#define NE_EDGES 400000

union FragBF { v16bf v; unsigned int u[8]; };

__device__ __forceinline__ float swishf(float x) { return x / (1.0f + __expf(-x)); }

#define GEMM_THREADS 256
#define GEMM_MTILE   128
#define APITCH       34
#define WPITCH       34
#define NT           7      // fixed: every block computes 7 n-tiles (weights zero-padded)

enum { EPI_STORE = 0, EPI_STATS = 1, EPI_SEGSUM = 2, EPI_RESID = 3, EPI_POOL = 4 };

// C(M x N) = swish-ish( A(M x K) @ Wt^T + bias ) with fused epilogues.
// A is bf16; Wt is bf16 stored transposed+padded as [NtileGroups*112][KPAD].
// GATHER==1: A row e = [x2bf[tgt[e]][0..299] , x2bf[src[e]][0..99], 0-pad]
template <int GATHER, int EPI>
__global__ __launch_bounds__(GEMM_THREADS) void gemm_bf16_wmma(
    const bf16_t* __restrict__ A, int Apitch, int Kact,
    const bf16_t* __restrict__ x2bf,
    const long long* __restrict__ srcI, const long long* __restrict__ tgtI,
    const bf16_t* __restrict__ Wt, int KPAD,
    const float* __restrict__ bias, int Nact, int M,
    bf16_t* __restrict__ outBf, int outPitch,
    float* __restrict__ colsum, float* __restrict__ colsumsq,
    float* __restrict__ segS,
    float* __restrict__ x2f, bf16_t* __restrict__ x2bfOut,
    float* __restrict__ pooled)
{
  __shared__ __align__(16) bf16_t Wl[NT * 16 * WPITCH];
  __shared__ __align__(16) bf16_t At[GEMM_MTILE * APITCH];
  __shared__ float ldsS[112];
  __shared__ float ldsQ[112];

  const int tid    = threadIdx.x;
  const int lane   = tid & 31;
  const int waveId = tid >> 5;
  const int mBase  = blockIdx.x * GEMM_MTILE;
  const int nt0    = blockIdx.y * NT;

  // staging assignment: thread -> (row, 16-col chunk) for A; strided for W
  const int srow  = tid >> 1;
  const int scol0 = (tid & 1) * 16;
  const int erow  = mBase + srow;
  long long gS = 0, gT = 0;
  if (GATHER) { if (erow < M) { gS = srcI[erow]; gT = tgtI[erow]; } }

  v8f acc[NT];
#pragma unroll
  for (int i = 0; i < NT; i++)
#pragma unroll
    for (int j = 0; j < 8; j++) acc[i][j] = 0.0f;

  // CDNA5 16-bit fragment lane maps (ISA 7.12.2)
  const int outer = lane & 15;
  const int kbA   = (lane >> 4) * 8;
  const int kbB   = (lane >> 4) * 8;
  int kA[8], kB[8];
#pragma unroll
  for (int j = 0; j < 8; j++) {
    kA[j] = ((j >> 2) * 16) + kbA + 2 * (j & 3);   // A: 16x32 bf16
    kB[j] = ((j >> 2) * 16) + kbB + 2 * (j & 3);   // B: 32x16 bf16
  }

  bf16_t aR[16];     // software-pipelined staging registers
  bf16_t wR[14];     // 112*32 / 256 threads = 14 elements

  auto loadA = [&](int k0) {
#pragma unroll
    for (int i = 0; i < 16; i++) {
      int gk = k0 + scol0 + i;
      bf16_t v = (bf16_t)0.0f;
      if (GATHER) {
        if (erow < M) {
          if (gk < 300)      v = x2bf[(size_t)gT * 300 + gk];
          else if (gk < 400) v = x2bf[(size_t)gS * 300 + (gk - 300)];
        }
      } else {
        if (erow < M && gk < Kact) v = A[(size_t)erow * Apitch + gk];
      }
      aR[i] = v;
    }
  };
  auto loadW = [&](int k0) {
#pragma unroll
    for (int i = 0; i < 14; i++) {
      int idx = tid + i * 256;          // covers 112*32 exactly
      int n = idx >> 5, k = idx & 31;
      wR[i] = Wt[(size_t)(nt0 * 16 + n) * KPAD + (k0 + k)];
    }
  };

  loadA(0);
  loadW(0);

  for (int k0 = 0; k0 < KPAD; k0 += 32) {
    // commit staged registers to LDS
#pragma unroll
    for (int i = 0; i < 16; i++) At[srow * APITCH + scol0 + i] = aR[i];
#pragma unroll
    for (int i = 0; i < 14; i++) {
      int idx = tid + i * 256;
      Wl[(idx >> 5) * WPITCH + (idx & 31)] = wR[i];
    }
    __syncthreads();

    // issue next-step global loads (overlap with WMMA below)
    int kn = k0 + 32;
    if (kn < KPAD) { loadA(kn); loadW(kn); }

    FragBF aF;
#pragma unroll
    for (int j = 0; j < 8; j++)
      aF.u[j] = *(const unsigned int*)&At[(waveId * 16 + outer) * APITCH + kA[j]];
#pragma unroll
    for (int nt = 0; nt < NT; nt++) {
      FragBF bF;
#pragma unroll
      for (int j = 0; j < 8; j++)
        bF.u[j] = *(const unsigned int*)&Wl[(nt * 16 + outer) * WPITCH + kB[j]];
      acc[nt] = __builtin_amdgcn_wmma_f32_16x16x32_bf16(
          false, aF.v, false, bF.v, (short)0, acc[nt], false, false);
    }
    __syncthreads();
  }

  if (EPI == EPI_STATS || EPI == EPI_POOL) {
    if (tid < 112) { ldsS[tid] = 0.0f; ldsQ[tid] = 0.0f; }
    __syncthreads();
  }

  const int mHalf = (lane >> 4) * 8;
  const int nloc  = lane & 15;
#pragma unroll
  for (int nt = 0; nt < NT; nt++) {
    int col = (nt0 + nt) * 16 + nloc;
    if (col >= Nact) continue;
    float bcol = bias[col];
#pragma unroll
    for (int j = 0; j < 8; j++) {
      int row = mBase + waveId * 16 + mHalf + j;
      if (row >= M) continue;
      float v = acc[nt][j] + bcol;
      if (EPI == EPI_POOL) {
        atomicAdd(&ldsS[nt * 16 + nloc], v);   // linear layer, pooled later
      } else {
        float h = swishf(v);
        if (EPI == EPI_STORE || EPI == EPI_STATS) {
          outBf[(size_t)row * outPitch + col] = (bf16_t)h;
          if (EPI == EPI_STATS) { atomicAdd(&ldsS[col], h); atomicAdd(&ldsQ[col], h * h); }
        } else if (EPI == EPI_SEGSUM) {
          long long t = tgtI[row];
          atomicAdd(&segS[(size_t)t * 100 + col], h);
        } else if (EPI == EPI_RESID) {
          size_t idx = (size_t)row * 300 + col;
          float nv = x2f[idx] + h;
          x2f[idx]     = nv;
          x2bfOut[idx] = (bf16_t)nv;
        }
      }
    }
  }

  if (EPI == EPI_STATS || EPI == EPI_POOL) {
    __syncthreads();
    if (tid < 112) {
      if (EPI == EPI_STATS) { atomicAdd(&colsum[tid], ldsS[tid]); atomicAdd(&colsumsq[tid], ldsQ[tid]); }
      else if (tid < Nact)  { atomicAdd(&pooled[tid], ldsS[tid]); }
    }
  }
}

// Fold batchnorm (from column stats) into next layer's weights:
// W'[n][k] = s_k * W[k][n] (bf16, [112][128]), b'[n] = b[n] + sum_k t_k W[k][n]
__global__ void fold_bn(const float* __restrict__ W, const float* __restrict__ b,
                        const float* __restrict__ gamma, const float* __restrict__ beta,
                        const float* __restrict__ colsum, const float* __restrict__ colsumsq,
                        float invM, bf16_t* __restrict__ WtOut, float* __restrict__ bOut)
{
  __shared__ float red[128];
  int n = blockIdx.x;     // 0..111
  int k = threadIdx.x;    // 0..127
  float s = 0.0f, t = 0.0f, w = 0.0f;
  if (k < 100) {
    float m   = colsum[k] * invM;
    float var = colsumsq[k] * invM - m * m;
    float sc  = gamma[k] * rsqrtf(var + 1e-5f);
    s = sc; t = beta[k] - m * sc;
    if (n < 100) w = W[k * 100 + n];
  }
  WtOut[n * 128 + k] = (bf16_t)(s * w);
  red[k] = t * w;
  __syncthreads();
  for (int st = 64; st > 0; st >>= 1) { if (k < st) red[k] += red[k + st]; __syncthreads(); }
  if (k == 0) bOut[n] = ((n < 100) ? b[n] : 0.0f) + red[0];
}

// transpose + pad f32 [K][N] -> bf16 [NPAD][KPAD]
__global__ void prep_w(const float* __restrict__ W, int K, int N, int KPAD, int total,
                       bf16_t* __restrict__ Wt)
{
  int idx = blockIdx.x * 256 + threadIdx.x;
  if (idx >= total) return;
  int n = idx / KPAD, k = idx - n * KPAD;
  float v = (k < K && n < N) ? W[k * N + n] : 0.0f;
  Wt[idx] = (bf16_t)v;
}

__global__ void init_x2(const float* __restrict__ nf, float* __restrict__ x2,
                        bf16_t* __restrict__ x2bf, int total)
{
  int i = blockIdx.x * 256 + threadIdx.x;
  if (i < total) { float v = nf[i]; x2[i] = v; x2bf[i] = (bf16_t)v; }
}

__global__ void zero_f32(float* __restrict__ p, int n)
{
  int i = blockIdx.x * 256 + threadIdx.x;
  if (i < n) p[i] = 0.0f;
}

__global__ void count_edges(const long long* __restrict__ tgtI, float* __restrict__ cnt, int NE)
{
  int e = blockIdx.x * 256 + threadIdx.x;
  if (e < NE) atomicAdd(&cnt[tgtI[e]], 1.0f);
}

// upd_in = concat([x1, x2, s/max(cnt,1)]) as bf16 [NN][512]
__global__ void build_updin(const float* __restrict__ nf, const float* __restrict__ x2,
                            const float* __restrict__ s, const float* __restrict__ cnt,
                            bf16_t* __restrict__ updin, int total)
{
  int idx = blockIdx.x * 256 + threadIdx.x;
  if (idx >= total) return;
  int n = idx >> 9, c = idx & 511;
  float v = 0.0f;
  if (c < 100)      v = nf[n * 300 + c];
  else if (c < 400) v = x2[n * 300 + (c - 100)];
  else if (c < 500) v = s[n * 100 + (c - 400)] / fmaxf(cnt[n], 1.0f);
  updin[idx] = (bf16_t)v;
}

__global__ void final_head(const float* __restrict__ pooled,
                           const float* __restrict__ W0, const float* __restrict__ b0,
                           const float* __restrict__ W1, const float* __restrict__ b1,
                           const float* __restrict__ nf, float* __restrict__ out)
{
  __shared__ float pm[100];
  __shared__ float c[100];
  int t = threadIdx.x;
  if (t < 100) pm[t] = pooled[t] * (1.0f / (float)NN_NODES);
  __syncthreads();
  if (t < 100) {
    float a = b0[t];
    for (int k = 0; k < 100; k++) a += pm[k] * W0[k * 100 + t];
    c[t] = swishf(a);
  }
  __syncthreads();
  if (t < 100) {
    float a = b1[t];
    for (int k = 0; k < 100; k++) a += c[k] * W1[k * 100 + t];
    out[t] = nf[t] * a;   // x1[0,:] = node_feature[0, :100]
  }
}

extern "C" void kernel_launch(void* const* d_in, const int* in_sizes, int n_in,
                              void* d_out, int out_size, void* d_ws, size_t ws_size,
                              hipStream_t stream)
{
  (void)in_sizes; (void)out_size; (void)ws_size;
  const int NN = NN_NODES, NE = NE_EDGES;

  const float*     nf   = (const float*)d_in[0];
  const long long* ei   = (const long long*)d_in[1];
  const long long* srcI = ei;
  const long long* tgtI = ei + NE;

  // params pytree leaf sizes (layers[0], layers[1], post, pre; dict keys sorted: bn,msg,upd)
  const int psz[56] = {
      100,100,100,100,100,100, 40000,100,10000,100,10000,100,10000,100,
      50000,100,10000,100,10000,100,30000,300,
      100,100,100,100,100,100, 40000,100,10000,100,10000,100,10000,100,
      50000,100,10000,100,10000,100,30000,300,
      10000,100,10000,100,
      30000,100,10000,100,10000,100,10000,100};
  auto P = [&](int i) -> const float* {
    if (n_in >= 59) return (const float*)d_in[3 + i];
    size_t off = 0; for (int j = 0; j < i; j++) off += (size_t)psz[j];
    return (const float*)d_in[3] + off;
  };

  // workspace carve-out
  size_t off = 0;
  auto alloc = [&](size_t bytes) -> void* {
    off = (off + 255) & ~(size_t)255;
    void* p = (char*)d_ws + off; off += bytes; return p;
  };
  float*  x2     = (float*) alloc((size_t)NN * 300 * 4);
  bf16_t* x2bf   = (bf16_t*)alloc((size_t)NN * 300 * 2);
  bf16_t* hA     = (bf16_t*)alloc((size_t)NE * 112 * 2);
  bf16_t* hB     = (bf16_t*)alloc((size_t)NE * 112 * 2);
  float*  segS   = (float*) alloc((size_t)NN * 100 * 4);
  float*  cnt    = (float*) alloc((size_t)NN * 4);
  bf16_t* updin  = (bf16_t*)alloc((size_t)NN * 512 * 2);
  float*  colsum = (float*) alloc(224 * 4);          // colsum[112] + colsumsq[112]
  float*  colsq  = colsum + 112;
  float*  pooled = (float*) alloc(112 * 4);
  bf16_t* fW     = (bf16_t*)alloc(112 * 128 * 2);
  float*  fB     = (float*) alloc(112 * 4);
  bf16_t* msgW1t[2], *updW0t[2], *updW1t[2], *updW2t[2], *updW3t[2];
  for (int l = 0; l < 2; l++) {
    msgW1t[l] = (bf16_t*)alloc(112 * 416 * 2);
    updW0t[l] = (bf16_t*)alloc(112 * 512 * 2);
    updW1t[l] = (bf16_t*)alloc(112 * 128 * 2);
    updW2t[l] = (bf16_t*)alloc(112 * 128 * 2);
    updW3t[l] = (bf16_t*)alloc(336 * 128 * 2);       // 21 n-tiles (3 groups of 7), zero-padded
  }
  bf16_t* preW0t = (bf16_t*)alloc(112 * 320 * 2);
  bf16_t* preW1t = (bf16_t*)alloc(112 * 128 * 2);
  bf16_t* preW2t = (bf16_t*)alloc(112 * 128 * 2);
  bf16_t* preW3t = (bf16_t*)alloc(112 * 128 * 2);

  auto cdiv = [](int a, int b) { return (a + b - 1) / b; };

  // ---- prep: x2 / x2bf init + weight transposes ----
  init_x2<<<cdiv(NN * 300, 256), 256, 0, stream>>>(nf, x2, x2bf, NN * 300);
  for (int l = 0; l < 2; l++) {
    int lb = l * 22;
    prep_w<<<cdiv(112 * 416, 256), 256, 0, stream>>>(P(lb + 6),  400, 100, 416, 112 * 416, msgW1t[l]);
    prep_w<<<cdiv(112 * 512, 256), 256, 0, stream>>>(P(lb + 14), 500, 100, 512, 112 * 512, updW0t[l]);
    prep_w<<<cdiv(112 * 128, 256), 256, 0, stream>>>(P(lb + 16), 100, 100, 128, 112 * 128, updW1t[l]);
    prep_w<<<cdiv(112 * 128, 256), 256, 0, stream>>>(P(lb + 18), 100, 100, 128, 112 * 128, updW2t[l]);
    prep_w<<<cdiv(336 * 128, 256), 256, 0, stream>>>(P(lb + 20), 100, 300, 128, 336 * 128, updW3t[l]);
  }
  prep_w<<<cdiv(112 * 320, 256), 256, 0, stream>>>(P(48), 300, 100, 320, 112 * 320, preW0t);
  prep_w<<<cdiv(112 * 128, 256), 256, 0, stream>>>(P(50), 100, 100, 128, 112 * 128, preW1t);
  prep_w<<<cdiv(112 * 128, 256), 256, 0, stream>>>(P(52), 100, 100, 128, 112 * 128, preW2t);
  prep_w<<<cdiv(112 * 128, 256), 256, 0, stream>>>(P(54), 100, 100, 128, 112 * 128, preW3t);

  const dim3 gE(NE / GEMM_MTILE, 1);            // 3125 blocks
  const dim3 gN(cdiv(NN, GEMM_MTILE), 1);       // 391 blocks
  const dim3 gN3(cdiv(NN, GEMM_MTILE), 3);      // N=300 -> 21 padded n-tiles over 3 y-blocks

  for (int l = 0; l < 2; l++) {
    int lb = l * 22;
    // msg L1: gathered edge features (K=400) -> hA, + bn0 stats
    zero_f32<<<1, 256, 0, stream>>>(colsum, 224);
    gemm_bf16_wmma<1, EPI_STATS><<<gE, GEMM_THREADS, 0, stream>>>(
        nullptr, 0, 0, x2bf, srcI, tgtI, msgW1t[l], 416, P(lb + 7), 100, NE,
        hA, 112, colsum, colsq, nullptr, nullptr, nullptr, nullptr);
    // fold bn0 into msg W2; msg L2 -> hB, + bn1 stats
    fold_bn<<<112, 128, 0, stream>>>(P(lb + 8), P(lb + 9), P(lb + 0), P(lb + 1),
                                     colsum, colsq, 1.0f / NE, fW, fB);
    zero_f32<<<1, 256, 0, stream>>>(colsum, 224);
    gemm_bf16_wmma<0, EPI_STATS><<<gE, GEMM_THREADS, 0, stream>>>(
        hA, 112, 100, nullptr, nullptr, nullptr, fW, 128, fB, 100, NE,
        hB, 112, colsum, colsq, nullptr, nullptr, nullptr, nullptr);
    // fold bn1 into msg W3; msg L3 -> hA, + bn2 stats
    fold_bn<<<112, 128, 0, stream>>>(P(lb + 10), P(lb + 11), P(lb + 2), P(lb + 3),
                                     colsum, colsq, 1.0f / NE, fW, fB);
    zero_f32<<<1, 256, 0, stream>>>(colsum, 224);
    gemm_bf16_wmma<0, EPI_STATS><<<gE, GEMM_THREADS, 0, stream>>>(
        hB, 112, 100, nullptr, nullptr, nullptr, fW, 128, fB, 100, NE,
        hA, 112, colsum, colsq, nullptr, nullptr, nullptr, nullptr);
    // fold bn2 into msg W4; msg L4 with fused segment-sum scatter
    fold_bn<<<112, 128, 0, stream>>>(P(lb + 12), P(lb + 13), P(lb + 4), P(lb + 5),
                                     colsum, colsq, 1.0f / NE, fW, fB);
    zero_f32<<<cdiv(NN * 100, 256), 256, 0, stream>>>(segS, NN * 100);
    zero_f32<<<cdiv(NN, 256), 256, 0, stream>>>(cnt, NN);
    gemm_bf16_wmma<0, EPI_SEGSUM><<<gE, GEMM_THREADS, 0, stream>>>(
        hA, 112, 100, nullptr, nullptr, tgtI, fW, 128, fB, 100, NE,
        nullptr, 0, nullptr, nullptr, segS, nullptr, nullptr, nullptr);
    count_edges<<<cdiv(NE, 256), 256, 0, stream>>>(tgtI, cnt, NE);
    // update net
    build_updin<<<cdiv(NN * 512, 256), 256, 0, stream>>>(nf, x2, segS, cnt, updin, NN * 512);
    gemm_bf16_wmma<0, EPI_STORE><<<gN, GEMM_THREADS, 0, stream>>>(
        updin, 512, 500, nullptr, nullptr, nullptr, updW0t[l], 512, P(lb + 15), 100, NN,
        hA, 112, nullptr, nullptr, nullptr, nullptr, nullptr, nullptr);
    gemm_bf16_wmma<0, EPI_STORE><<<gN, GEMM_THREADS, 0, stream>>>(
        hA, 112, 100, nullptr, nullptr, nullptr, updW1t[l], 128, P(lb + 17), 100, NN,
        hB, 112, nullptr, nullptr, nullptr, nullptr, nullptr, nullptr);
    gemm_bf16_wmma<0, EPI_STORE><<<gN, GEMM_THREADS, 0, stream>>>(
        hB, 112, 100, nullptr, nullptr, nullptr, updW2t[l], 128, P(lb + 19), 100, NN,
        hA, 112, nullptr, nullptr, nullptr, nullptr, nullptr, nullptr);
    // final update layer: residual into x2 (f32) + refresh bf16 copy
    gemm_bf16_wmma<0, EPI_RESID><<<gN3, GEMM_THREADS, 0, stream>>>(
        hA, 112, 100, nullptr, nullptr, nullptr, updW3t[l], 128, P(lb + 21), 300, NN,
        nullptr, 0, nullptr, nullptr, nullptr, x2, x2bf, nullptr);
  }

  // pre-net head
  gemm_bf16_wmma<0, EPI_STORE><<<gN, GEMM_THREADS, 0, stream>>>(
      x2bf, 300, 300, nullptr, nullptr, nullptr, preW0t, 320, P(49), 100, NN,
      hA, 112, nullptr, nullptr, nullptr, nullptr, nullptr, nullptr);
  gemm_bf16_wmma<0, EPI_STORE><<<gN, GEMM_THREADS, 0, stream>>>(
      hA, 112, 100, nullptr, nullptr, nullptr, preW1t, 128, P(51), 100, NN,
      hB, 112, nullptr, nullptr, nullptr, nullptr, nullptr, nullptr);
  gemm_bf16_wmma<0, EPI_STORE><<<gN, GEMM_THREADS, 0, stream>>>(
      hB, 112, 100, nullptr, nullptr, nullptr, preW2t, 128, P(53), 100, NN,
      hA, 112, nullptr, nullptr, nullptr, nullptr, nullptr, nullptr);
  zero_f32<<<1, 256, 0, stream>>>(pooled, 112);
  gemm_bf16_wmma<0, EPI_POOL><<<gN, GEMM_THREADS, 0, stream>>>(
      hA, 112, 100, nullptr, nullptr, nullptr, preW3t, 128, P(55), 100, NN,
      nullptr, 0, nullptr, nullptr, nullptr, nullptr, nullptr, pooled);

  final_head<<<1, 128, 0, stream>>>(pooled, P(44), P(45), P(46), P(47), nf, (float*)d_out);
}
